// SummaryForgetAttention_68624987456193
// MI455X (gfx1250) — matrix operand order repaired and checked by reading, not compiled, not run on hardware
//
#include <hip/hip_runtime.h>
#include <hip/hip_bf16.h>

// SummaryForgetAttention for MI455X (gfx1250), wave32 + WMMA f16.
// Pipeline:
//   convertT (Wk,Wv,Wo -> f16, K-major)       : one-shot layout pass
//   gemm<1,f32> keys  @ WkT -> UK (row-major) : gate, sigmoid
//   gemm<1,f32> values@ WvT -> UV (row-major)
//   fill summary halves of UK / UV
//   attn_kernel: flash attention, S computed TRANSPOSED (K*Q^T) so softmax
//                rows are in-lane; streams attn_weights/mask as b128/b64
//   gemm<0,f16> AO @ WoT -> out (fp32 + bias)

typedef __attribute__((ext_vector_type(16))) _Float16 v16h;
typedef __attribute__((ext_vector_type(8)))  _Float16 v8h;
typedef __attribute__((ext_vector_type(8)))  float    v8f;

#define D_MODEL 1024
#define NSEQ    2048
#define NHEAD   16
#define DHEAD   64

__device__ __forceinline__ v8f wmma_f16(v16h a, v16h b, v8f c) {
  return __builtin_amdgcn_wmma_f32_16x16x32_f16(
      false, a, false, b, (short)0, c, false, false);
}

__device__ __forceinline__ v16h merge8(v8h lo, v8h hi) {
  return __builtin_shufflevector(lo, hi, 0, 1, 2, 3, 4, 5, 6, 7,
                                 8, 9, 10, 11, 12, 13, 14, 15);
}

// B fragment: column gather, rows p[0..7] and p[16..23] (stride ld).
__device__ __forceinline__ v16h ld_frag16(const _Float16* p, int ld) {
  v16h r;
#pragma unroll
  for (int i = 0; i < 16; ++i) r[i] = p[(size_t)(i < 8 ? i : i + 8) * ld];
  return r;
}

// A fragment, f16 source: runs p[0..7] and p[16..23].
__device__ __forceinline__ v16h a_frag(const _Float16* p) {
  return merge8(*reinterpret_cast<const v8h*>(p),
                *reinterpret_cast<const v8h*>(p + 16));
}

// A fragment, fp32 source: runs p[0..7] and p[16..23].
__device__ __forceinline__ v16h a_frag_f32(const float* p) {
  float4 f0 = *reinterpret_cast<const float4*>(p);
  float4 f1 = *reinterpret_cast<const float4*>(p + 4);
  float4 f2 = *reinterpret_cast<const float4*>(p + 16);
  float4 f3 = *reinterpret_cast<const float4*>(p + 20);
  float t[16] = {f0.x, f0.y, f0.z, f0.w, f1.x, f1.y, f1.z, f1.w,
                 f2.x, f2.y, f2.z, f2.w, f3.x, f3.y, f3.z, f3.w};
  v16h a;
#pragma unroll
  for (int i = 0; i < 16; ++i) a[i] = (_Float16)t[i];
  return a;
}

__device__ __forceinline__ v16h a_frag_any(const float* p) { return a_frag_f32(p); }
__device__ __forceinline__ v16h a_frag_any(const _Float16* p) { return a_frag(p); }

// ---------------------------------------------------------------------------
// One-shot transpose+convert: W [N][K] fp32 -> Wt [K][N] f16.
// ---------------------------------------------------------------------------
__global__ void convertT(const float* __restrict__ W, _Float16* __restrict__ Wt,
                         int N, int K) {
  const int idx = blockIdx.x * blockDim.x + threadIdx.x;
  if (idx >= N * K) return;
  const int n = idx / K;
  const int k = idx - n * K;
  Wt[(size_t)k * N + n] = (_Float16)W[idx];
}

// ---------------------------------------------------------------------------
// C = A(MxK row-major, fp32 or f16) * Bt(KxN f16, K-major) + epilogue.
// Block = 128 threads (4 waves) -> 64x64 tile; wave -> 16x64.
// MODE 0: outF[m][n] = acc + bias1[n]                      (fp32)
// MODE 1: outH[m][n] = f16(sigmoid(acc+bias1[n]+bias2[n]))
// ---------------------------------------------------------------------------
template <int MODE, typename AT>
__global__ __launch_bounds__(128) void gemm_nt_wmma(
    const AT* __restrict__ A, const _Float16* __restrict__ Bt,
    const float* __restrict__ bias1, const float* __restrict__ bias2,
    float* __restrict__ outF, _Float16* __restrict__ outH,
    int K, int ldb, int ldo) {
  const int lane   = threadIdx.x & 31;
  const int wave   = threadIdx.x >> 5;
  const int half16 = lane >> 4;
  const int l16    = lane & 15;
  const int n0 = blockIdx.x * 64;
  const int m0 = blockIdx.y * 64 + wave * 16;

  v8f acc[4] = {};
  const AT* ap = A + (size_t)(m0 + l16) * K + half16 * 8;
  const _Float16* bp = Bt + (size_t)(8 * half16) * ldb + n0 + l16;

  for (int kb = 0; kb < K; kb += 32) {
    v16h a = a_frag_any(ap);
#pragma unroll
    for (int t = 0; t < 4; ++t) {
      v16h bf = ld_frag16(bp + 16 * t, ldb);
      acc[t] = wmma_f16(a, bf, acc[t]);
    }
    ap += 32;
    bp += (size_t)32 * ldb;
  }

#pragma unroll
  for (int t = 0; t < 4; ++t) {
#pragma unroll
    for (int j = 0; j < 8; ++j) {
      const int m = m0 + j + 8 * half16;
      const int n = n0 + 16 * t + l16;
      float v = acc[t][j];
      if (MODE == 0) {
        outF[(size_t)m * ldo + n] = v + bias1[n];
      } else {
        v = v + bias1[n] + bias2[n];
        v = 1.0f / (1.0f + __expf(-v));
        outH[(size_t)m * ldo + n] = (_Float16)v;
      }
    }
  }
}

// ---------------------------------------------------------------------------
// Summary broadcast: UK[:,512+j]=k_sum[j], UV[:,512+j]=v_sum[j].
// ---------------------------------------------------------------------------
__global__ void fill_summary(const float* __restrict__ k_sum,
                             const float* __restrict__ v_sum,
                             _Float16* __restrict__ UK,
                             _Float16* __restrict__ UV) {
  const int idx = blockIdx.x * blockDim.x + threadIdx.x;
  if (idx >= 2 * NSEQ * (D_MODEL / 2)) return;
  const int j = idx & 511;
  const size_t row = (size_t)(idx >> 9);
  UK[row * D_MODEL + 512 + j] = (_Float16)k_sum[j];
  UV[row * D_MODEL + 512 + j] = (_Float16)v_sum[j];
}

// ---------------------------------------------------------------------------
// Flash attention, transposed scores: block = 256 threads (8 waves) handles
// (b, h, 128 q rows); each wave owns 16 queries, loops over 64-key tiles.
// S^T = K*Q^T puts each query's keys in-lane -> softmax without lane shuffles
// (one xor-16 combine), vectorized W (b128) / mask (b64) / P (ds b128) access.
// ---------------------------------------------------------------------------
#define KT_LD 72   // f16 stride (144B, 16B-aligned, bank-spread)
#define VT_LD 72
#define PT_LD 72

__global__ __launch_bounds__(256, 1) void attn_kernel(
    const float* __restrict__ Q, const _Float16* __restrict__ UK,
    const _Float16* __restrict__ UV, const unsigned char* __restrict__ mask,
    const float* __restrict__ W, _Float16* __restrict__ AO) {
  __shared__ alignas(16) _Float16 Qt[8][DHEAD][16];   // per-wave Q, d-major
  __shared__ alignas(16) _Float16 Kt[64][KT_LD];      // [key][d] 64x64 tile
  __shared__ alignas(16) _Float16 Vt[64][VT_LD];      // [key][d] 64x64 tile
  __shared__ alignas(16) _Float16 Pt[8][16][PT_LD];   // per-wave P^T->[query][key]
  __shared__ alignas(16) float    Sred[8][16];        // per-wave alpha / lsum

  const int lane   = threadIdx.x & 31;
  const int wave   = threadIdx.x >> 5;
  const int half16 = lane >> 4;
  const int l16    = lane & 15;
  const int h  = blockIdx.y;
  const int b  = blockIdx.z;
  const int qw = blockIdx.x * 128 + wave * 16;

  // Stage this wave's Q (16 queries x 64 dims) d-major, scaled by 1/sqrt(64).
  {
    const float* qp = Q + ((size_t)(b * NSEQ + qw + l16)) * D_MODEL + h * DHEAD
                      + 32 * half16;
#pragma unroll
    for (int i = 0; i < 8; ++i) {
      float4 f = *reinterpret_cast<const float4*>(qp + 4 * i);
      const int d = 32 * half16 + 4 * i;
      Qt[wave][d + 0][l16] = (_Float16)(f.x * 0.125f);
      Qt[wave][d + 1][l16] = (_Float16)(f.y * 0.125f);
      Qt[wave][d + 2][l16] = (_Float16)(f.z * 0.125f);
      Qt[wave][d + 3][l16] = (_Float16)(f.w * 0.125f);
    }
  }
  asm volatile("s_wait_dscnt 0" ::: "memory");
  // Loop-invariant Q B-fragments (column = this lane's query, elements over d).
  v16h qb[2];
  qb[0] = ld_frag16(&Qt[wave][8 * half16][l16], 16);
  qb[1] = ld_frag16(&Qt[wave][32 + 8 * half16][l16], 16);

  v8f acc[4] = {};
  float mcur = -1e30f, lsum = 0.0f;   // per-lane = per-query state

  // Per-lane streams: query row = qw+l16; key runs of 8 at 16s + 8*half16.
  const size_t srow = ((size_t)(b * NHEAD + h) * NSEQ + qw + l16) * NSEQ + 8 * half16;
  const float*         Wq = W + srow;
  const unsigned char* Mq = mask + srow;

  // K/V staging: 16 f16 (two float4) per thread per tile.
  const int trow = threadIdx.x >> 2;           // 0..63 (key)
  const int tcol = (threadIdx.x & 3) * 16;     // 0,16,32,48 (d)
  const _Float16* ksrc = UK + ((size_t)(b * NSEQ + trow)) * D_MODEL + h * DHEAD + tcol;
  const _Float16* vsrc = UV + ((size_t)(b * NSEQ + trow)) * D_MODEL + h * DHEAD + tcol;

  for (int kb = 0; kb < NSEQ; kb += 64) {
    __syncthreads();
    {
      const float4* sk = reinterpret_cast<const float4*>(ksrc + (size_t)kb * D_MODEL);
      float4* dk = reinterpret_cast<float4*>(&Kt[trow][tcol]);
      dk[0] = sk[0]; dk[1] = sk[1];
      const float4* sv = reinterpret_cast<const float4*>(vsrc + (size_t)kb * D_MODEL);
      float4* dv = reinterpret_cast<float4*>(&Vt[trow][tcol]);
      dv[0] = sv[0]; dv[1] = sv[1];
    }
    __syncthreads();

    // S^T (64 keys x 16 queries): A = K-tile rows, B = Q (loop-invariant).
    v8f sc[4];
#pragma unroll
    for (int s = 0; s < 4; ++s) {
      v8f z = {};
#pragma unroll
      for (int kk = 0; kk < 2; ++kk) {
        v16h ka = a_frag(&Kt[16 * s + l16][kk * 32 + 8 * half16]);
        z = wmma_f16(ka, qb[kk], z);
      }
      sc[s] = z;  // sc[s][j] = S[key = kb+16s+j+8*half16][query = qw+l16]
    }

    // Apply attn_weights (b128 loads) and mask (b64 loads) in place.
#pragma unroll
    for (int s = 0; s < 4; ++s) {
      float4 wA = *reinterpret_cast<const float4*>(Wq + 16 * s);
      float4 wB = *reinterpret_cast<const float4*>(Wq + 16 * s + 4);
      uint2  mm = *reinterpret_cast<const uint2*>(Mq + 16 * s);
      const float wv[8] = {wA.x, wA.y, wA.z, wA.w, wB.x, wB.y, wB.z, wB.w};
#pragma unroll
      for (int j = 0; j < 8; ++j) {
        const unsigned byte =
            ((j < 4 ? mm.x : mm.y) >> (8 * (j & 3))) & 0xffu;
        sc[s][j] = byte ? -1e30f : sc[s][j] * wv[j];
      }
    }

    // In-lane max over 32 keys + one xor-16 combine (this query's tile max).
    float tmax = -1e30f;
#pragma unroll
    for (int s = 0; s < 4; ++s)
#pragma unroll
      for (int j = 0; j < 8; ++j) tmax = fmaxf(tmax, sc[s][j]);
    tmax = fmaxf(tmax, __shfl_xor(tmax, 16, 32));

    const float mnew  = fmaxf(mcur, tmax);
    const float alpha = __expf(mcur - mnew);
    mcur = mnew;

    // Exponentiate in place; in-lane sum + one xor-16 combine.
    float rsum = 0.0f;
#pragma unroll
    for (int s = 0; s < 4; ++s)
#pragma unroll
      for (int j = 0; j < 8; ++j) {
        sc[s][j] = __expf(sc[s][j] - mnew);
        rsum += sc[s][j];
      }
    rsum += __shfl_xor(rsum, 16, 32);
    lsum = lsum * alpha + rsum;

    // P^T -> per-wave LDS as [query][key]: contiguous 8-key runs (b128 stores).
#pragma unroll
    for (int s = 0; s < 4; ++s) {
      v8h pe;
#pragma unroll
      for (int j = 0; j < 8; ++j) pe[j] = (_Float16)sc[s][j];
      *reinterpret_cast<v8h*>(&Pt[wave][l16][16 * s + 8 * half16]) = pe;
    }
    Sred[wave][l16] = alpha;            // redistribute alpha to acc-row layout
    asm volatile("s_wait_dscnt 0" ::: "memory");

    // Rescale accumulator rows (queries j+8*half16) by their alphas.
    {
      float4 a0 = *reinterpret_cast<const float4*>(&Sred[wave][8 * half16]);
      float4 a1 = *reinterpret_cast<const float4*>(&Sred[wave][8 * half16 + 4]);
      const float af[8] = {a0.x, a0.y, a0.z, a0.w, a1.x, a1.y, a1.z, a1.w};
#pragma unroll
      for (int t = 0; t < 4; ++t)
#pragma unroll
        for (int j = 0; j < 8; ++j) acc[t][j] *= af[j];
    }

    // acc(16q x 64d) += P(16q x 64k) * V(64k x 64d).
    const int vk = 8 * half16;
#pragma unroll
    for (int c = 0; c < 2; ++c) {
      v16h pa = merge8(
          *reinterpret_cast<const v8h*>(&Pt[wave][l16][32 * c + half16 * 8]),
          *reinterpret_cast<const v8h*>(&Pt[wave][l16][32 * c + 16 + half16 * 8]));
#pragma unroll
      for (int t = 0; t < 4; ++t) {
        v16h bf = ld_frag16(&Vt[32 * c + vk][16 * t + l16], VT_LD);
        acc[t] = wmma_f16(pa, bf, acc[t]);
      }
    }

    __builtin_prefetch(Wq + 64, 0, 0);  // next attn_weights block
    Wq += 64;
    Mq += 64;
  }

  // Normalize: redistribute per-query 1/lsum to acc-row layout, store f16.
  Sred[wave][l16] = lsum;
  asm volatile("s_wait_dscnt 0" ::: "memory");
  {
    float4 l0 = *reinterpret_cast<const float4*>(&Sred[wave][8 * half16]);
    float4 l1 = *reinterpret_cast<const float4*>(&Sred[wave][8 * half16 + 4]);
    float li[8] = {l0.x, l0.y, l0.z, l0.w, l1.x, l1.y, l1.z, l1.w};
#pragma unroll
    for (int j = 0; j < 8; ++j) li[j] = 1.0f / li[j];
#pragma unroll
    for (int t = 0; t < 4; ++t)
#pragma unroll
      for (int j = 0; j < 8; ++j) {
        const int q = qw + j + 8 * half16;
        const int n = 16 * t + l16;
        AO[((size_t)(b * NSEQ + q) * D_MODEL) + h * DHEAD + n] =
            (_Float16)(acc[t][j] * li[j]);
      }
  }
}

// ---------------------------------------------------------------------------
extern "C" void kernel_launch(void* const* d_in, const int* in_sizes, int n_in,
                              void* d_out, int out_size, void* d_ws, size_t ws_size,
                              hipStream_t stream) {
  const float*         Qf   = (const float*)d_in[0];
  const float*         Kf   = (const float*)d_in[1];
  const float*         Vf   = (const float*)d_in[2];
  const unsigned char* Msk  = (const unsigned char*)d_in[3];
  const float*         Wat  = (const float*)d_in[4];
  const float*         Wk   = (const float*)d_in[5];
  const float*         bk   = (const float*)d_in[6];
  const float*         kfb  = (const float*)d_in[7];
  const float*         Wv   = (const float*)d_in[8];
  const float*         bv   = (const float*)d_in[9];
  const float*         vfb  = (const float*)d_in[10];
  const float*         ksum = (const float*)d_in[11];
  const float*         vsum = (const float*)d_in[12];
  const float*         Wo   = (const float*)d_in[13];
  const float*         bo   = (const float*)d_in[14];
  float* out = (float*)d_out;

  // Workspace layout (MiB offsets)
  char* ws = (char*)d_ws;
  _Float16* UK  = (_Float16*)ws;                               //  8 MiB [4096][1024]
  _Float16* UV  = (_Float16*)(ws + 8u  * 1024 * 1024);         //  8 MiB [4096][1024]
  _Float16* AO  = (_Float16*)(ws + 16u * 1024 * 1024);         //  8 MiB [4096][1024]
  _Float16* WkT = (_Float16*)(ws + 24u * 1024 * 1024);         //  1 MiB [1024][512]
  _Float16* WvT = (_Float16*)(ws + 25u * 1024 * 1024);         //  1 MiB [1024][512]
  _Float16* WoT = (_Float16*)(ws + 26u * 1024 * 1024);         //  2 MiB [1024][1024]
  (void)ws_size; (void)in_sizes; (void)n_in; (void)out_size;

  const dim3 blk(128);

  // One-shot weight transpose+convert to f16 K-major.
  convertT<<<(512 * 1024 + 255) / 256, 256, 0, stream>>>(Wk, WkT, 512, D_MODEL);
  convertT<<<(512 * 1024 + 255) / 256, 256, 0, stream>>>(Wv, WvT, 512, D_MODEL);
  convertT<<<(1024 * 1024 + 255) / 256, 256, 0, stream>>>(Wo, WoT, 1024, D_MODEL);

  // Gate linears + sigmoid -> row-major UK/UV. M=4096, N=512, K=1024.
  gemm_nt_wmma<1, float><<<dim3(8, 64), blk, 0, stream>>>(
      Kf, WkT, bk, kfb, nullptr, UK, D_MODEL, 512, D_MODEL);
  gemm_nt_wmma<1, float><<<dim3(8, 64), blk, 0, stream>>>(
      Vf, WvT, bv, vfb, nullptr, UV, D_MODEL, 512, D_MODEL);

  // Summary halves.
  fill_summary<<<(2 * NSEQ * 512 + 255) / 256, 256, 0, stream>>>(ksum, vsum, UK, UV);

  // Flash attention: grid (NQ/128, H, B), 256 threads.
  attn_kernel<<<dim3(NSEQ / 128, NHEAD, 2), 256, 0, stream>>>(
      Qf, UK, UV, Msk, Wat, AO);

  // Output projection: out = AO(f16) @ Wo^T + bo. M=4096, N=1024, K=1024.
  gemm_nt_wmma<0, _Float16><<<dim3(16, 64), blk, 0, stream>>>(
      AO, WoT, bo, nullptr, out, nullptr, D_MODEL, 1024, D_MODEL);
}
